// ViTSelfAttention_24163486007353
// MI455X (gfx1250) — matrix operand order repaired and checked by reading, not compiled, not run on hardware
//
#include <hip/hip_runtime.h>
#include <hip/hip_bf16.h>
#include <stdint.h>

#define N_ROWS 32768
#define M_COLS 2048
#define DIM    128
#define HEADS  8
#define LSCALE 0.08838834764831845f   // 1/sqrt(128)

typedef __attribute__((ext_vector_type(16))) __bf16        v16bf;
typedef __attribute__((ext_vector_type(8)))  float         v8f;
typedef __attribute__((ext_vector_type(4)))  unsigned int  u32x4;

union FragBF { v16bf v; u32x4 q[2]; };

__device__ __forceinline__ unsigned short f2bf(float f) {
  unsigned int u = __float_as_uint(f);
  u += 0x7FFFu + ((u >> 16) & 1u);          // round-to-nearest-even
  return (unsigned short)(u >> 16);
}
__device__ __forceinline__ float bf2f(unsigned short h) {
  return __uint_as_float(((unsigned int)h) << 16);
}

// ---- Kernel 1a: fold heads into weights: Wsum = sum_h Wq_h (bf16), bsum = 2*sum_h bq_h
__global__ __launch_bounds__(128) void prep_w_kernel(const float* __restrict__ Wq,
                                                     const float* __restrict__ bq,
                                                     unsigned short* __restrict__ Wsumb,
                                                     float* __restrict__ bsum) {
  int i = blockIdx.x * blockDim.x + threadIdx.x;     // 0..16383
  int d = i >> 7, j = i & 127;
  float s = 0.f;
  #pragma unroll
  for (int h = 0; h < HEADS; ++h) s += Wq[(h * DIM + d) * DIM + j];
  Wsumb[d * DIM + j] = f2bf(s);
  if (j == 0) {
    float b = 0.f;
    #pragma unroll
    for (int h = 0; h < HEADS; ++h) b += bq[h * DIM + d];
    bsum[d] = 2.0f * b;
  }
}

// ---- Kernel 1b: c -> bf16
__global__ __launch_bounds__(256) void prep_c_kernel(const float* __restrict__ c,
                                                     unsigned short* __restrict__ cb) {
  int i = blockIdx.x * blockDim.x + threadIdx.x;     // 0..M*DIM-1
  cb[i] = f2bf(c[i]);
}

// ---- Kernel 2: tsum = (x1 + k) @ Wsum.T + bsum, stored bf16 [N, DIM]
#define TS_ROWS 32
__global__ __launch_bounds__(128) void tsum_kernel(const float* __restrict__ x1,
                                                   const float* __restrict__ kk,
                                                   const unsigned short* __restrict__ Wsumb,
                                                   const float* __restrict__ bsum,
                                                   unsigned short* __restrict__ tsumb) {
  __shared__ float sRow[TS_ROWS * DIM];
  const int tid = threadIdx.x;                       // 0..127 -> output column d
  const int n0  = blockIdx.x * TS_ROWS;
  for (int e = tid; e < TS_ROWS * DIM; e += 128) {
    int g = n0 * DIM + e;
    sRow[e] = x1[g] + kk[g];
  }
  __syncthreads();
  float acc[TS_ROWS];
  #pragma unroll
  for (int r = 0; r < TS_ROWS; ++r) acc[r] = 0.f;
  for (int j = 0; j < DIM; ++j) {
    float w = bf2f(Wsumb[tid * DIM + j]);
    #pragma unroll
    for (int r = 0; r < TS_ROWS; ++r) acc[r] = fmaf(sRow[r * DIM + j], w, acc[r]);
  }
  float b = bsum[tid];
  #pragma unroll
  for (int r = 0; r < TS_ROWS; ++r)
    tsumb[(size_t)(n0 + r) * DIM + tid] = f2bf(acc[r] + b);
}

// ---- Kernel 3: fused logits GEMM (bf16 WMMA) + softmax Z + (p+gumbel) argmax.
// Each wave owns 32 rows (two 16-row A fragment sets reusing each B fragment twice).
// Softmax is shift-invariant and |logit| <~ 10 here, so exp() needs no max-shift:
// pass 1 accumulates Z = sum exp(x) directly; pass 2 recomputes x and argmaxes
// p + gumbel (cheaper than materializing the 268 MB [N,M] logits in HBM).
__global__ __launch_bounds__(256) void attn_argmax_kernel(const unsigned short* __restrict__ tsumb,
                                                          const unsigned short* __restrict__ cb,
                                                          int* __restrict__ idx_out) {
  const int lane = threadIdx.x & 31;
  const int wave = threadIdx.x >> 5;
  const int r0   = (blockIdx.x * 8 + wave) * 32;
  const int half = lane >> 4;        // lane group (ISA 16-bit frag layout)
  const int col  = lane & 15;        // A row / B,D column within tile

  // A fragments: tsum rows r0..r0+31, K=0..127 (resident for both passes)
  FragBF a[2][4];
  #pragma unroll
  for (int g = 0; g < 2; ++g)
    #pragma unroll
    for (int f = 0; f < 4; ++f) {
      const unsigned short* base =
          tsumb + (size_t)(r0 + g * 16 + col) * DIM + f * 32 + half * 16;
      a[g][f].q[0] = *(const u32x4*)(base);        // K = 32f + 16*half + 0..7
      a[g][f].q[1] = *(const u32x4*)(base + 8);    // K = 32f + 16*half + 8..15
    }

  float zsum[2][8];
  #pragma unroll
  for (int g = 0; g < 2; ++g)
    #pragma unroll
    for (int v = 0; v < 8; ++v) zsum[g][v] = 0.f;

  // ---- Pass 1: Z = sum_m exp(logit * scale)
  for (int m0 = 0; m0 < M_COLS; m0 += 16) {
    __builtin_prefetch(cb + (size_t)(m0 + 16 + col) * DIM, 0, 0);  // global_prefetch_b8
    FragBF b[4];
    #pragma unroll
    for (int f = 0; f < 4; ++f) {
      const unsigned short* bp = cb + (size_t)(m0 + col) * DIM + f * 32 + half * 16;
      b[f].q[0] = *(const u32x4*)(bp);             // K = 32f + 16*half + 0..7
      b[f].q[1] = *(const u32x4*)(bp + 8);         // K = 32f + 16*half + 8..15
    }
    v8f acc0 = {}, acc1 = {};
    #pragma unroll
    for (int f = 0; f < 4; ++f)
      acc0 = __builtin_amdgcn_wmma_f32_16x16x32_bf16(false, a[0][f].v, false, b[f].v,
                                                     (short)0, acc0, false, false);
    #pragma unroll
    for (int f = 0; f < 4; ++f)
      acc1 = __builtin_amdgcn_wmma_f32_16x16x32_bf16(false, a[1][f].v, false, b[f].v,
                                                     (short)0, acc1, false, false);
    #pragma unroll
    for (int v = 0; v < 8; ++v) zsum[0][v] += __expf(acc0[v] * LSCALE);
    #pragma unroll
    for (int v = 0; v < 8; ++v) zsum[1][v] += __expf(acc1[v] * LSCALE);
  }
  // sum across the 16 column-lanes of each half (masks<16 stay inside each half)
  #pragma unroll
  for (int mask = 1; mask < 16; mask <<= 1)
    #pragma unroll
    for (int g = 0; g < 2; ++g)
      #pragma unroll
      for (int v = 0; v < 8; ++v) zsum[g][v] += __shfl_xor(zsum[g][v], mask, 32);

  float rinv[2][8];
  #pragma unroll
  for (int g = 0; g < 2; ++g)
    #pragma unroll
    for (int v = 0; v < 8; ++v) rinv[g][v] = 1.0f / zsum[g][v];

  // loop-invariant row component of the gumbel hash
  unsigned int nh[2][8];
  #pragma unroll
  for (int g = 0; g < 2; ++g)
    #pragma unroll
    for (int v = 0; v < 8; ++v)
      nh[g][v] = ((unsigned)(r0 + g * 16 + half * 8 + v) * 0x9E3779B9u) ^ 42u;

  float best[2][8]; int bidx[2][8];
  #pragma unroll
  for (int g = 0; g < 2; ++g)
    #pragma unroll
    for (int v = 0; v < 8; ++v) { best[g][v] = -3.0e38f; bidx[g][v] = 0; }

  // ---- Pass 2: recompute logits, p = exp(x)/Z, argmax(p + gumbel)
  for (int m0 = 0; m0 < M_COLS; m0 += 16) {
    FragBF b[4];
    #pragma unroll
    for (int f = 0; f < 4; ++f) {
      const unsigned short* bp = cb + (size_t)(m0 + col) * DIM + f * 32 + half * 16;
      b[f].q[0] = *(const u32x4*)(bp);
      b[f].q[1] = *(const u32x4*)(bp + 8);
    }
    v8f acc0 = {}, acc1 = {};
    #pragma unroll
    for (int f = 0; f < 4; ++f)
      acc0 = __builtin_amdgcn_wmma_f32_16x16x32_bf16(false, a[0][f].v, false, b[f].v,
                                                     (short)0, acc0, false, false);
    #pragma unroll
    for (int f = 0; f < 4; ++f)
      acc1 = __builtin_amdgcn_wmma_f32_16x16x32_bf16(false, a[1][f].v, false, b[f].v,
                                                     (short)0, acc1, false, false);
    const int m = m0 + col;
    const unsigned int mh = (unsigned)m * 0x85EBCA6Bu;   // shared across all rows this lane holds
    #pragma unroll
    for (int g = 0; g < 2; ++g)
      #pragma unroll
      for (int v = 0; v < 8; ++v) {
        float x = (g ? acc1[v] : acc0[v]) * LSCALE;
        float p = __expf(x) * rinv[g][v];
        unsigned int s = nh[g][v] ^ mh;
        s ^= s >> 16; s *= 0x7FEB352Du;
        s ^= s >> 15; s *= 0x846CA68Bu;
        s ^= s >> 16;
        float u  = (float)(s >> 8) * (0.99999988f / 16777216.0f) + 1.1754944e-7f;
        float sc = p - __logf(-__logf(u));               // p + Gumbel(0,1)
        if (sc > best[g][v]) { best[g][v] = sc; bidx[g][v] = m; }
      }
  }
  // cross-lane argmax (tie-break: lower index, matching jnp.argmax)
  #pragma unroll
  for (int mask = 1; mask < 16; mask <<= 1)
    #pragma unroll
    for (int g = 0; g < 2; ++g)
      #pragma unroll
      for (int v = 0; v < 8; ++v) {
        float ob = __shfl_xor(best[g][v], mask, 32);
        int   oi = __shfl_xor(bidx[g][v], mask, 32);
        if (ob > best[g][v] || (ob == best[g][v] && oi < bidx[g][v])) {
          best[g][v] = ob; bidx[g][v] = oi;
        }
      }
  if (col == 0) {
    #pragma unroll
    for (int g = 0; g < 2; ++g)
      #pragma unroll
      for (int v = 0; v < 8; ++v)
        idx_out[r0 + g * 16 + half * 8 + v] = bidx[g][v];
  }
}

// ---- Kernel 4a/4b: out_acc = one_hot(idx).T @ x1 via scatter-add
__global__ __launch_bounds__(256) void zero_kernel(float* __restrict__ p, int n) {
  int i = blockIdx.x * blockDim.x + threadIdx.x;
  if (i < n) p[i] = 0.f;
}
__global__ __launch_bounds__(256) void scatter_kernel(const float* __restrict__ x1,
                                                      const int* __restrict__ idx,
                                                      float* __restrict__ out_acc) {
  int g = blockIdx.x * blockDim.x + threadIdx.x;   // N*DIM
  int n = g >> 7, d = g & 127;
  atomicAdd(&out_acc[(size_t)idx[n] * DIM + d], x1[g]);
}

// ---- Kernel 5: out = out_acc @ Wd.T + bd  (tiny, 67 MFLOP)
__global__ __launch_bounds__(256) void final_kernel(const float* __restrict__ out_acc,
                                                    const float* __restrict__ Wd,
                                                    const float* __restrict__ bd,
                                                    float* __restrict__ out) {
  int g = blockIdx.x * blockDim.x + threadIdx.x;   // M*DIM
  int m = g >> 7, dp = g & 127;
  float acc = bd[dp];
  const float* arow = out_acc + (size_t)m * DIM;
  const float* wrow = Wd + (size_t)dp * DIM;
  #pragma unroll 8
  for (int j = 0; j < DIM; ++j) acc = fmaf(arow[j], wrow[j], acc);
  out[g] = acc;
}

extern "C" void kernel_launch(void* const* d_in, const int* in_sizes, int n_in,
                              void* d_out, int out_size, void* d_ws, size_t ws_size,
                              hipStream_t stream) {
  const float* x1 = (const float*)d_in[0];
  const float* k  = (const float*)d_in[1];
  const float* c  = (const float*)d_in[2];
  const float* Wq = (const float*)d_in[3];
  const float* bq = (const float*)d_in[4];
  const float* Wd = (const float*)d_in[5];
  const float* bd = (const float*)d_in[6];
  float* out = (float*)d_out;
  (void)in_sizes; (void)n_in; (void)out_size; (void)ws_size;

  char* ws = (char*)d_ws;
  size_t off = 0;
  auto wsalloc = [&](size_t bytes) -> void* {
    void* p = ws + off;
    off = (off + bytes + 255) & ~(size_t)255;
    return p;
  };
  unsigned short* cb    = (unsigned short*)wsalloc((size_t)M_COLS * DIM * 2);  // 512 KB
  unsigned short* Wsumb = (unsigned short*)wsalloc((size_t)DIM * DIM * 2);     // 32 KB
  float*          bsum  = (float*)wsalloc((size_t)DIM * 4);
  unsigned short* tsumb = (unsigned short*)wsalloc((size_t)N_ROWS * DIM * 2);  // 8 MB
  int*            idx   = (int*)wsalloc((size_t)N_ROWS * 4);                   // 128 KB
  float*          oacc  = (float*)wsalloc((size_t)M_COLS * DIM * 4);           // 1 MB

  prep_w_kernel<<<128, 128, 0, stream>>>(Wq, bq, Wsumb, bsum);
  prep_c_kernel<<<(M_COLS * DIM) / 256, 256, 0, stream>>>(c, cb);
  tsum_kernel<<<N_ROWS / TS_ROWS, 128, 0, stream>>>(x1, k, Wsumb, bsum, tsumb);
  attn_argmax_kernel<<<N_ROWS / 256, 256, 0, stream>>>(tsumb, cb, idx);
  zero_kernel<<<(M_COLS * DIM) / 256, 256, 0, stream>>>(oacc, M_COLS * DIM);
  scatter_kernel<<<(N_ROWS * DIM) / 256, 256, 0, stream>>>(x1, idx, oacc);
  final_kernel<<<(M_COLS * DIM) / 256, 256, 0, stream>>>(oacc, Wd, bd, out);
}
